// MultiHeadAttentionLayer_2997887173089
// MI455X (gfx1250) — matrix operand order repaired and checked by reading, not compiled, not run on hardware
//
#include <hip/hip_runtime.h>

// Sizes fixed by the reference.
#define B_   2
#define S_   4096
#define H_   512
#define NH_  8
#define HD_  64
#define MTOT (B_ * S_)          // 8192 rows for the projection GEMMs

typedef __attribute__((ext_vector_type(16))) __bf16 v16bf;
typedef __attribute__((ext_vector_type(8)))  __bf16 v8bf;
typedef __attribute__((ext_vector_type(4)))  __bf16 v4bf;
typedef __attribute__((ext_vector_type(8)))  float  v8f;
typedef __attribute__((ext_vector_type(4)))  float  v4f;
typedef __attribute__((ext_vector_type(4)))  unsigned int u32x4;
typedef __attribute__((ext_vector_type(8)))  int i32x8;
typedef __attribute__((ext_vector_type(4)))  int i32x4;

#define WMMA_BF16(acc, a, b)                                                   \
  (acc) = __builtin_amdgcn_wmma_f32_16x16x32_bf16(false, (a), false, (b),      \
                                                  (short)0, (acc), false, false)

// ---------------------------------------------------------------------------
// Tensor Data Mover: 2D tile (bf16 elements) global -> LDS.
// D# packing per CDNA5 ISA 8.3/8.4:
//  group0: [1:0]=count=1, [63:32]=lds_addr, [120:64]=global_addr, [127:126]=2
//  group1: [17:16]=data_size(1 => 2B), tensor_dim0/1, tile_dim0/1,
//          tensor_dim0_stride; everything else (mask/barrier/pad/iter) = 0.
// Issued by one wave; completion via TENSORcnt.
// This toolchain exposes the 6-arg builtin: (g0, g1, g2, g3, g_extra, cpol).
// ---------------------------------------------------------------------------
static __device__ __forceinline__ void tdm_load_2d_bf16(
    unsigned lds_off_bytes, const void* gptr,
    unsigned tile_d0, unsigned tile_d1,
    unsigned long long row_stride_elems) {
  unsigned long long ga = (unsigned long long)gptr;
  u32x4 g0;
  g0[0] = 1u;                                            // count=1
  g0[1] = lds_off_bytes;                                 // lds_addr
  g0[2] = (unsigned)(ga & 0xffffffffull);                // global_addr lo
  g0[3] = (unsigned)((ga >> 32) & 0x1ffffffull) | (2u << 30);  // hi | type=2
  unsigned td0 = tile_d0, td1 = tile_d1;                 // exact dims: no clip
  i32x8 g1;
  g1[0] = (int)(1u << 16);                               // data_size=2B
  g1[1] = (int)((td0 & 0xffffu) << 16);                  // tensor_dim0 lo16
  g1[2] = (int)((td0 >> 16) | ((td1 & 0xffffu) << 16));  // td0 hi | td1 lo
  g1[3] = (int)((td1 >> 16) | (tile_d0 << 16));          // td1 hi | tile_dim0
  g1[4] = (int)(tile_d1 & 0xffffu);                      // tile_dim1, tile_dim2=0
  g1[5] = (int)(row_stride_elems & 0xffffffffull);       // dim0_stride lo32
  g1[6] = (int)((row_stride_elems >> 32) & 0xffffull);   // dim0_stride hi16
  g1[7] = 0;
  i32x4 z4 = {0, 0, 0, 0};
  i32x8 z8 = {0, 0, 0, 0, 0, 0, 0, 0};
  __builtin_amdgcn_tensor_load_to_lds(g0, g1, z4, z4, z8, 0);
}

// ---------------------------------------------------------------------------
// Fragment loaders, per CDNA5 ISA 7.12.2 (wave32) VGPR layouts.
// ---------------------------------------------------------------------------
static __device__ __forceinline__ v16bf frag_a_bf16(const __bf16* __restrict__ tile,
                                                    int stride) {
  const int lane = threadIdx.x & 31;
  const int m    = lane & 15;
  const int half = lane >> 4;
  const __bf16* p = tile + (size_t)m * stride + half * 8;
  v8bf lo = *(const v8bf*)(p);
  v8bf hi = *(const v8bf*)(p + 16);
  v16bf f;
#pragma unroll
  for (int i = 0; i < 8; ++i) { f[i] = lo[i]; f[i + 8] = hi[i]; }
  return f;
}

// B[k][n] = src[n][k], src N-major with row stride `stride`.
static __device__ __forceinline__ v16bf frag_b_bf16(const __bf16* __restrict__ tile,
                                                    int stride) {
  const int lane = threadIdx.x & 31;
  const int n    = lane & 15;
  const int half = lane >> 4;
  const __bf16* p = tile + (size_t)n * stride + half * 8;
  v8bf lo = *(const v8bf*)(p);
  v8bf hi = *(const v8bf*)(p + 16);
  v16bf f;
#pragma unroll
  for (int i = 0; i < 8; ++i) { f[i] = lo[i]; f[i + 8] = hi[i]; }
  return f;
}

// A fragment sourced from fp32 (softmax probabilities), converted on the fly.
static __device__ __forceinline__ v16bf frag_a_f32(const float* __restrict__ tile,
                                                   int stride) {
  const int lane = threadIdx.x & 31;
  const int m    = lane & 15;
  const int half = lane >> 4;
  const float* p = tile + (size_t)m * stride + half * 8;
  v4f a0 = *(const v4f*)(p);
  v4f a1 = *(const v4f*)(p + 4);
  v4f b0 = *(const v4f*)(p + 16);
  v4f b1 = *(const v4f*)(p + 20);
  v16bf f;
#pragma unroll
  for (int i = 0; i < 4; ++i) {
    f[i]      = (__bf16)a0[i];
    f[i + 4]  = (__bf16)a1[i];
    f[i + 8]  = (__bf16)b0[i];
    f[i + 12] = (__bf16)b1[i];
  }
  return f;
}

// ---------------------------------------------------------------------------
// fp32 -> bf16 bulk conversion.
// ---------------------------------------------------------------------------
__global__ __launch_bounds__(256)
void cvt_f32_bf16(const float* __restrict__ src, __bf16* __restrict__ dst, int n4) {
  int i = blockIdx.x * 256 + threadIdx.x;
  if (i < n4) {
    v4f v = ((const v4f*)src)[i];
    v4bf o;
    o[0] = (__bf16)v[0]; o[1] = (__bf16)v[1];
    o[2] = (__bf16)v[2]; o[3] = (__bf16)v[3];
    ((v4bf*)dst)[i] = o;
  }
}

// ---------------------------------------------------------------------------
// GEMM:  out[m][n] = sum_k X[m][k] * W[n][k] + bias[n]      (x @ W^T + b)
// WG tile 128x128; wave tile 32x64.
// MODE 0: bf16 head-major [B,NH,S,HD]; MODE 1: bf16 [B,NH,HD,S]; MODE 2: fp32.
// ---------------------------------------------------------------------------
template <int MODE>
__global__ __launch_bounds__(256)
void gemm_xWt(const __bf16* __restrict__ X, const __bf16* __restrict__ W,
              const float* __restrict__ bias, void* __restrict__ out, int K) {
  const int wave = threadIdx.x >> 5;
  const int lane = threadIdx.x & 31;
  const int wm = wave & 3, wn = wave >> 2;
  const int m0 = blockIdx.x * 128 + wm * 32;
  const int n0 = blockIdx.y * 128 + wn * 64;

  v8f acc[2][4] = {};

  for (int k = 0; k < K; k += 32) {
    __builtin_prefetch(X + (size_t)m0 * K + k + 128, 0, 1);
    v16bf a0 = frag_a_bf16(X + (size_t)m0 * K + k, K);
    v16bf a1 = frag_a_bf16(X + (size_t)(m0 + 16) * K + k, K);
#pragma unroll
    for (int j = 0; j < 4; ++j) {
      v16bf b = frag_b_bf16(W + (size_t)(n0 + 16 * j) * K + k, K);
      WMMA_BF16(acc[0][j], a0, b);
      WMMA_BF16(acc[1][j], a1, b);
    }
  }

  const int half = lane >> 4;
  const int col  = lane & 15;
#pragma unroll
  for (int i = 0; i < 2; ++i)
#pragma unroll
    for (int j = 0; j < 4; ++j)
#pragma unroll
      for (int r = 0; r < 8; ++r) {
        int m = m0 + i * 16 + half * 8 + r;
        int n = n0 + j * 16 + col;
        float v = acc[i][j][r] + bias[n];
        if (MODE == 2) {
          ((float*)out)[(size_t)m * H_ + n] = v;
        } else {
          int bb = m >> 12, ss = m & (S_ - 1);
          int hh = n >> 6,  dd = n & (HD_ - 1);
          if (MODE == 0)
            ((__bf16*)out)[((size_t)(bb * NH_ + hh) * S_ + ss) * HD_ + dd] = (__bf16)v;
          else
            ((__bf16*)out)[((size_t)(bb * NH_ + hh) * HD_ + dd) * S_ + ss] = (__bf16)v;
        }
      }
}

// ---------------------------------------------------------------------------
// energy[bh][q][j] = 0.125 * sum_d Q[q][d] K[j][d].
// Q-tile (128x64) and K-tile (128x64) staged in LDS by the TDM (contiguous
// 16KB each), then both are reused by all 8 waves from LDS (ds_load_b128).
// Dynamic LDS: 32KB.  Q at offset 0, K at offset 16384.
// ---------------------------------------------------------------------------
__global__ __launch_bounds__(256)
void attn_energy(const __bf16* __restrict__ Qh, const __bf16* __restrict__ Kh,
                 float* __restrict__ energy) {
  extern __shared__ __bf16 lds[];        // [0,8192): Qtile, [8192,16384): Ktile
  __bf16* Qt = lds;
  __bf16* Kt = lds + 128 * HD_;

  const int bh = blockIdx.z;
  const int wave = threadIdx.x >> 5;
  const int lane = threadIdx.x & 31;
  const int wm = wave & 3, wn = wave >> 2;
  const int qblk = blockIdx.x * 128;
  const int jblk = blockIdx.y * 128;

  if (wave == 0) {
    const __bf16* Qg = Qh + (size_t)bh * S_ * HD_ + (size_t)qblk * HD_;
    const __bf16* Kg = Kh + (size_t)bh * S_ * HD_ + (size_t)jblk * HD_;
    tdm_load_2d_bf16(0u,     Qg, 128 * HD_, 1, 128 * HD_);   // contiguous 16KB
    tdm_load_2d_bf16(16384u, Kg, 128 * HD_, 1, 128 * HD_);
    __builtin_amdgcn_s_wait_tensorcnt(0);
  }
  __syncthreads();

  const int q0 = wm * 32;     // within LDS tile
  const int j0 = wn * 64;

  v8f acc[2][4] = {};
#pragma unroll
  for (int k = 0; k < HD_; k += 32) {
    v16bf a0 = frag_a_bf16(Qt + (size_t)q0 * HD_ + k, HD_);
    v16bf a1 = frag_a_bf16(Qt + (size_t)(q0 + 16) * HD_ + k, HD_);
#pragma unroll
    for (int j = 0; j < 4; ++j) {
      v16bf b = frag_b_bf16(Kt + (size_t)(j0 + 16 * j) * HD_ + k, HD_);
      WMMA_BF16(acc[0][j], a0, b);
      WMMA_BF16(acc[1][j], a1, b);
    }
  }

  float* E = energy + (size_t)bh * S_ * S_;
  const float scale = 0.125f;  // 1/sqrt(64)
  const int half = lane >> 4;
  const int col  = lane & 15;
#pragma unroll
  for (int i = 0; i < 2; ++i)
#pragma unroll
    for (int j = 0; j < 4; ++j)
#pragma unroll
      for (int r = 0; r < 8; ++r) {
        int qq = qblk + q0 + i * 16 + half * 8 + r;
        int jj = jblk + j0 + j * 16 + col;
        E[(size_t)qq * S_ + jj] = acc[i][j][r] * scale;
      }
}

// ---------------------------------------------------------------------------
// In-place row softmax over 4096-wide rows.
// ---------------------------------------------------------------------------
__global__ __launch_bounds__(256)
void softmax_rows(float* __restrict__ attn) {
  __shared__ float smax[8];
  __shared__ float ssum[8];
  float* p = attn + (size_t)blockIdx.x * S_;
  const int tid = threadIdx.x;

  v4f v[4];
  float m = -3.402823466e38f;
#pragma unroll
  for (int i = 0; i < 4; ++i) {
    v[i] = ((const v4f*)p)[tid + 256 * i];
    m = fmaxf(m, fmaxf(fmaxf(v[i][0], v[i][1]), fmaxf(v[i][2], v[i][3])));
  }
#pragma unroll
  for (int off = 16; off > 0; off >>= 1) m = fmaxf(m, __shfl_xor(m, off, 32));
  if ((tid & 31) == 0) smax[tid >> 5] = m;
  __syncthreads();
  float rowmax = smax[0];
#pragma unroll
  for (int i = 1; i < 8; ++i) rowmax = fmaxf(rowmax, smax[i]);

  float s = 0.f;
#pragma unroll
  for (int i = 0; i < 4; ++i) {
    v[i][0] = __expf(v[i][0] - rowmax);
    v[i][1] = __expf(v[i][1] - rowmax);
    v[i][2] = __expf(v[i][2] - rowmax);
    v[i][3] = __expf(v[i][3] - rowmax);
    s += v[i][0] + v[i][1] + v[i][2] + v[i][3];
  }
#pragma unroll
  for (int off = 16; off > 0; off >>= 1) s += __shfl_xor(s, off, 32);
  if ((tid & 31) == 0) ssum[tid >> 5] = s;
  __syncthreads();
  float tot = 0.f;
#pragma unroll
  for (int i = 0; i < 8; ++i) tot += ssum[i];
  float inv = 1.0f / tot;
#pragma unroll
  for (int i = 0; i < 4; ++i) {
    v[i][0] *= inv; v[i][1] *= inv; v[i][2] *= inv; v[i][3] *= inv;
    ((v4f*)p)[tid + 256 * i] = v[i];
  }
}

// ---------------------------------------------------------------------------
// Xc[b][q][h*HD+d] = sum_j P[bh][q][j] * V[bh][j][d],  V stored [bh][d][j].
// Wave tiling: 8 waves x 16 rows (q) each, full 64-wide d -> every P element
// is read exactly once per WG (the dominant 1.07GB stream).
// V is TDM-staged in 64x256 bf16 chunks (32KB), double buffered: the TDM for
// chunk c+1 runs while WMMAs consume chunk c.  Dynamic LDS: 64KB.
// ---------------------------------------------------------------------------
__global__ __launch_bounds__(256)
void attn_pv(const float* __restrict__ attn, const __bf16* __restrict__ Vt,
             __bf16* __restrict__ Xc) {
  extern __shared__ __bf16 vbuf[];       // two 64x256 chunks: 16384 elems each
  const int bh = blockIdx.z;
  const int b = bh >> 3, h = bh & 7;
  const float* P  = attn + (size_t)bh * S_ * S_;
  const __bf16* V = Vt + (size_t)bh * HD_ * S_;   // [HD][S]

  const int wave = threadIdx.x >> 5;
  const int lane = threadIdx.x & 31;
  const int q0 = blockIdx.x * 128 + wave * 16;    // 16 rows per wave

  const int CHUNK = 256;                           // k-elements per LDS chunk
  const int NCH   = S_ / CHUNK;                    // 16 chunks

  if (wave == 0)
    tdm_load_2d_bf16(0u, V, CHUNK, HD_, S_);       // chunk 0 -> buffer 0

  v8f acc[4] = {};
  for (int c = 0; c < NCH; ++c) {
    if (wave == 0) __builtin_amdgcn_s_wait_tensorcnt(0);  // chunk c resident
    __syncthreads();
    if (wave == 0 && c + 1 < NCH)                  // overlap next chunk DMA
      tdm_load_2d_bf16((unsigned)(((c + 1) & 1) * CHUNK * HD_ * 2),
                       V + (size_t)(c + 1) * CHUNK, CHUNK, HD_, S_);

    const __bf16* buf = vbuf + (size_t)(c & 1) * CHUNK * HD_;
    const int k0 = c * CHUNK;
#pragma unroll
    for (int kk = 0; kk < CHUNK; kk += 32) {
      __builtin_prefetch(P + (size_t)q0 * S_ + k0 + kk + 256, 0, 1);
      v16bf a = frag_a_f32(P + (size_t)q0 * S_ + k0 + kk, S_);
#pragma unroll
      for (int j = 0; j < 4; ++j) {
        v16bf bf = frag_b_bf16(buf + (size_t)(16 * j) * CHUNK + kk, CHUNK);
        WMMA_BF16(acc[j], a, bf);
      }
    }
    __syncthreads();   // all waves done with buffer (c&1) before it's refilled
  }

  const int half = lane >> 4;
  const int col  = lane & 15;
#pragma unroll
  for (int j = 0; j < 4; ++j)
#pragma unroll
    for (int r = 0; r < 8; ++r) {
      int qq = q0 + half * 8 + r;
      int dd = j * 16 + col;
      Xc[((size_t)b * S_ + qq) * H_ + h * HD_ + dd] = (__bf16)acc[j][r];
    }
}

// ---------------------------------------------------------------------------
extern "C" void kernel_launch(void* const* d_in, const int* in_sizes, int n_in,
                              void* d_out, int out_size, void* d_ws, size_t ws_size,
                              hipStream_t stream) {
  (void)in_sizes; (void)n_in; (void)out_size; (void)ws_size;
  const float* q_in = (const float*)d_in[0];
  const float* k_in = (const float*)d_in[1];
  const float* v_in = (const float*)d_in[2];
  const float* Wq = (const float*)d_in[3];
  const float* bq = (const float*)d_in[4];
  const float* Wk = (const float*)d_in[5];
  const float* bk = (const float*)d_in[6];
  const float* Wv = (const float*)d_in[7];
  const float* bv = (const float*)d_in[8];
  const float* Wo = (const float*)d_in[9];
  const float* bo = (const float*)d_in[10];

  const size_t XE = (size_t)MTOT * H_;  // 4,194,304 activation elements
  const size_t WE = (size_t)H_ * H_;    // 262,144 weight elements

  __bf16* ws = (__bf16*)d_ws;
  __bf16* Xq  = ws;               // [MTOT][H]
  __bf16* Xk  = Xq  + XE;
  __bf16* Xv  = Xk  + XE;
  __bf16* Wqb = Xv  + XE;         // [H][H]
  __bf16* Wkb = Wqb + WE;
  __bf16* Wvb = Wkb + WE;
  __bf16* Wob = Wvb + WE;
  __bf16* Qh  = Wob + WE;         // [B,NH,S,HD]
  __bf16* Kh  = Qh  + XE;
  __bf16* Vt  = Kh  + XE;         // [B,NH,HD,S]
  __bf16* Xc  = Vt  + XE;         // [B,S,H]

  float* x_out = (float*)d_out;                 // [B,S,H]
  float* attn  = x_out + XE;                    // [B,NH,S,S]

  // 1) fp32 -> bf16 conversions.
  cvt_f32_bf16<<<(int)(XE / 1024), 256, 0, stream>>>(q_in, Xq, (int)(XE / 4));
  cvt_f32_bf16<<<(int)(XE / 1024), 256, 0, stream>>>(k_in, Xk, (int)(XE / 4));
  cvt_f32_bf16<<<(int)(XE / 1024), 256, 0, stream>>>(v_in, Xv, (int)(XE / 4));
  cvt_f32_bf16<<<(int)(WE / 1024), 256, 0, stream>>>(Wq, Wqb, (int)(WE / 4));
  cvt_f32_bf16<<<(int)(WE / 1024), 256, 0, stream>>>(Wk, Wkb, (int)(WE / 4));
  cvt_f32_bf16<<<(int)(WE / 1024), 256, 0, stream>>>(Wv, Wvb, (int)(WE / 4));
  cvt_f32_bf16<<<(int)(WE / 1024), 256, 0, stream>>>(Wo, Wob, (int)(WE / 4));

  // 2) Q/K/V projections (WMMA bf16, fp32 accum).
  dim3 gproj(MTOT / 128, H_ / 128, 1);
  gemm_xWt<0><<<gproj, 256, 0, stream>>>(Xq, Wqb, bq, Qh, H_);
  gemm_xWt<0><<<gproj, 256, 0, stream>>>(Xk, Wkb, bk, Kh, H_);
  gemm_xWt<1><<<gproj, 256, 0, stream>>>(Xv, Wvb, bv, Vt, H_);

  // 3) energy = scale * Q K^T (TDM-staged tiles), fp32 into d_out slot.
  attn_energy<<<dim3(S_ / 128, S_ / 128, B_ * NH_), 256, 32768, stream>>>(Qh, Kh, attn);

  // 4) row softmax, in place in d_out.
  softmax_rows<<<B_ * NH_ * S_, 256, 0, stream>>>(attn);

  // 5) context = P V  (TDM double-buffered V, unique P reads).
  attn_pv<<<dim3(S_ / 128, 1, B_ * NH_), 256, 65536, stream>>>(attn, Vt, Xc);

  // 6) x = Xc @ Wo^T + bo, fp32 into d_out.
  gemm_xWt<2><<<gproj, 256, 0, stream>>>(Xc, Wob, bo, x_out, H_);
}
